// TranslatorEncoderRNN_10874857194115
// MI455X (gfx1250) — matrix operand order repaired and checked by reading, not compiled, not run on hardware
//
#include <hip/hip_runtime.h>
#include <hip/hip_bf16.h>

#define B_   64
#define T_   512
#define E_   512
#define H_   512
#define G4_  2048   // 4*H
#define KTOT 1024   // E + H

typedef __attribute__((ext_vector_type(16))) __bf16 v16bf;
typedef __attribute__((ext_vector_type(8)))  __bf16 v8bf;
typedef __attribute__((ext_vector_type(8)))  float  v8f;

static __device__ __forceinline__ float sigf(float x) {
    return __builtin_amdgcn_rcpf(1.0f + __expf(-x));   // v_rcp_f32, no IEEE div
}
static __device__ __forceinline__ float tanh_fast(float x) {
    x = fminf(fmaxf(x, -15.0f), 15.0f);
    return 1.0f - 2.0f * __builtin_amdgcn_rcpf(1.0f + __expf(2.0f * x));
}
// Pack two contiguous 8x-bf16 (16B) chunks into a 16-element WMMA fragment.
static __device__ __forceinline__ v16bf pack16(const __bf16* p0, const __bf16* p1) {
    v8bf lo = *(const v8bf*)p0;
    v8bf hi = *(const v8bf*)p1;
    v16bf r;
#pragma unroll
    for (int i = 0; i < 8; ++i) { r[i] = lo[i]; r[8 + i] = hi[i]; }
    return r;
}

// ---------------- prep kernels ----------------

__global__ void embed_kernel(const int* __restrict__ inputs,
                             const float* __restrict__ emb,
                             __bf16* __restrict__ X) {
    int idx = blockIdx.x * blockDim.x + threadIdx.x;     // T*B*(E/8) threads
    if (idx >= T_ * B_ * (E_ / 8)) return;
    int k8  = idx & 63;        // E/8 = 64
    int row = idx >> 6;        // t*B + b
    int t   = row >> 6;        // B = 64
    int b   = row & 63;
    int tok = inputs[b * T_ + t];
    v8bf o;
    if (tok == 0) {
#pragma unroll
        for (int i = 0; i < 8; ++i) o[i] = (__bf16)0.0f;
    } else {
        const float* p = emb + (size_t)tok * E_ + k8 * 8;
        float4 f0 = *(const float4*)p;
        float4 f1 = *(const float4*)(p + 4);
        o[0] = (__bf16)f0.x; o[1] = (__bf16)f0.y; o[2] = (__bf16)f0.z; o[3] = (__bf16)f0.w;
        o[4] = (__bf16)f1.x; o[5] = (__bf16)f1.y; o[6] = (__bf16)f1.z; o[7] = (__bf16)f1.w;
    }
    *(v8bf*)(X + (size_t)row * E_ + k8 * 8) = o;
}

__global__ void mask_kernel(const int* __restrict__ lengths,
                            float* __restrict__ maskT,       // (T,B)
                            float* __restrict__ mask_out) {  // (B,T) -> d_out
    int idx = blockIdx.x * blockDim.x + threadIdx.x;         // B*T
    if (idx >= B_ * T_) return;
    int b = idx >> 9;     // T = 512
    int t = idx & 511;
    float m = (t < lengths[b]) ? 1.0f : 0.0f;
    maskT[t * B_ + b]    = m;
    mask_out[b * T_ + t] = m;
}

// Wc[dir][n][k] = bf16( k<E ? Wih[n][k] : Whh[n][k-E] ); (N,K) row-major = WMMA B layout.
__global__ void wcomb_kernel(const float* __restrict__ WihF, const float* __restrict__ WhhF,
                             const float* __restrict__ WihB, const float* __restrict__ WhhB,
                             __bf16* __restrict__ Wc) {
    size_t idx = (size_t)blockIdx.x * blockDim.x + threadIdx.x;   // 2*2048*1024
    if (idx >= (size_t)2 * G4_ * KTOT) return;
    int k   = (int)(idx & (KTOT - 1));
    int n   = (int)((idx >> 10) & (G4_ - 1));
    int dir = (int)(idx >> 21);
    const float* Wih = dir ? WihB : WihF;
    const float* Whh = dir ? WhhB : WhhF;
    float v = (k < E_) ? Wih[(size_t)n * E_ + k] : Whh[(size_t)n * H_ + (k - E_)];
    Wc[idx] = (__bf16)v;
}

__global__ void bias_kernel(const float* __restrict__ bihF, const float* __restrict__ bhhF,
                            const float* __restrict__ bihB, const float* __restrict__ bhhB,
                            float* __restrict__ bias) {
    int idx = blockIdx.x * blockDim.x + threadIdx.x;   // 2*2048
    if (idx >= 2 * G4_) return;
    int n = idx & (G4_ - 1);
    int dir = idx >> 11;
    bias[idx] = dir ? (bihB[n] + bhhB[n]) : (bihF[n] + bhhF[n]);
}

__global__ void zero_f32(float* __restrict__ p, int n) {
    int i = blockIdx.x * blockDim.x + threadIdx.x;
    if (i < n) p[i] = 0.0f;
}
__global__ void zero_bf16(__bf16* __restrict__ p, int n) {
    int i = blockIdx.x * blockDim.x + threadIdx.x;
    if (i < n) p[i] = (__bf16)0.0f;
}

// ---------------- recurrent step ----------------
// Grid: (16, 2) blocks of 256 threads (8 waves). blockIdx.y = direction.
// Wave w: mi = w&3 -> batch tile rows [16*mi,16*mi+16); jb = w>>2 -> hidden col block.
// Each wave accumulates the SAME 16 hidden columns for all 4 gates -> LSTM cell
// update is fully lane-local (no LDS, no cross-wave traffic).
struct Frags { v16bf a, b0, b1, b2, b3; };

__global__ __launch_bounds__(256) void lstm_step_kernel(
    int s,
    const __bf16* __restrict__ X,        // (T,B,E) bf16
    const float*  __restrict__ maskT,    // (T,B)
    const __bf16* __restrict__ Wc,       // (2,4H,E+H) bf16
    const float*  __restrict__ bias,     // (2,4H)
    const __bf16* __restrict__ hbf_r,    // (2,B,H) bf16   read state
    __bf16*       __restrict__ hbf_w,    // (2,B,H) bf16   write state
    const float*  __restrict__ hf_r,     // (2,B,H) f32    read state
    float*        __restrict__ hf_w,     // (2,B,H) f32    write state
    float*        __restrict__ c,        // (2,B,H) f32    in-place
    float*        __restrict__ out)      // (B,T,2H) f32 -> d_out
{
    const int dir  = blockIdx.y;
    const int t    = dir ? (T_ - 1 - s) : s;
    const int lane = threadIdx.x & 31;
    const int wave = threadIdx.x >> 5;
    const int n16  = lane & 15;
    const int hh   = lane >> 4;
    const int mi   = wave & 3;
    const int jb   = wave >> 2;
    const int j    = blockIdx.x * 32 + jb * 16 + n16;   // hidden column 0..511
    const int m    = mi * 16 + n16;                     // A-fragment row (batch)

    v8f acc[4];
#pragma unroll
    for (int g = 0; g < 4; ++g) {
        float bv = bias[dir * G4_ + g * H_ + j];
        v8f a;
#pragma unroll
        for (int r = 0; r < 8; ++r) a[r] = bv;
        acc[g] = a;
    }

    // Fixed base pointers; all loads below are base + compile-time immediate offset.
    const __bf16* xrow = X     + ((size_t)t * B_ + m) * E_ + 16 * hh;
    const __bf16* hrow = hbf_r + ((size_t)dir * B_ + m) * H_ + 16 * hh;
    const __bf16* wrow = Wc    + ((size_t)dir * G4_ + j) * KTOT + 16 * hh;  // gate 0
    const size_t GSTRIDE = (size_t)H_ * KTOT;   // gate-to-gate stride in Wc

    // Unified K loop: iteration kt covers K = [kt*32, kt*32+32).
    // kt < 16 -> A from x_t (offset kt*32); kt >= 16 -> A from h (offset kt*32-E).
    // Weight offset is kt*32 in both regions. All selects resolve at compile time.
    auto loadfrags = [&](int kt) -> Frags {
        Frags f;
        const __bf16* ap = (kt < 16) ? (xrow + kt * 32) : (hrow + (kt * 32 - E_));
        f.a = pack16(ap, ap + 8);
        const __bf16* p0 = wrow + 0 * GSTRIDE + kt * 32;
        const __bf16* p1 = wrow + 1 * GSTRIDE + kt * 32;
        const __bf16* p2 = wrow + 2 * GSTRIDE + kt * 32;
        const __bf16* p3 = wrow + 3 * GSTRIDE + kt * 32;
        f.b0 = pack16(p0, p0 + 8);
        f.b1 = pack16(p1, p1 + 8);
        f.b2 = pack16(p2, p2 + 8);
        f.b3 = pack16(p3, p3 + 8);
        return f;
    };

    // Software pipeline, prefetch distance 2: loads for kt+2 are issued before
    // the WMMAs of kt, so fragment live ranges overlap across WMMAs and the
    // scheduler can use partial s_wait_loadcnt instead of full drains.
    Frags cur = loadfrags(0);
    Frags nxt = loadfrags(1);
#pragma unroll
    for (int kt = 0; kt < KTOT / 32; ++kt) {
        Frags nn = cur;                       // harmless default for tail iters
        if (kt + 2 < KTOT / 32) nn = loadfrags(kt + 2);
        acc[0] = __builtin_amdgcn_wmma_f32_16x16x32_bf16(false, cur.a, false, cur.b0, (short)0, acc[0], false, false);
        acc[1] = __builtin_amdgcn_wmma_f32_16x16x32_bf16(false, cur.a, false, cur.b1, (short)0, acc[1], false, false);
        acc[2] = __builtin_amdgcn_wmma_f32_16x16x32_bf16(false, cur.a, false, cur.b2, (short)0, acc[2], false, false);
        acc[3] = __builtin_amdgcn_wmma_f32_16x16x32_bf16(false, cur.a, false, cur.b3, (short)0, acc[3], false, false);
        cur = nxt;
        nxt = nn;
    }

    // LSTM cell update: C element r of this lane = (row mi*16 + 8*hh + r, col j).
#pragma unroll
    for (int r = 0; r < 8; ++r) {
        int   brow = mi * 16 + 8 * hh + r;
        float mval = maskT[t * B_ + brow];
        float iv = sigf(acc[0][r]);
        float fv = sigf(acc[1][r]);
        float gv = tanh_fast(acc[2][r]);
        float ov = sigf(acc[3][r]);
        size_t sidx = ((size_t)dir * B_ + brow) * H_ + j;
        float cold = c[sidx];
        float hold = hf_r[sidx];
        float cnew = fv * cold + iv * gv;
        float hnew = ov * tanh_fast(cnew);
        float hb = fmaf(mval, hnew - hold, hold);   // mask blend as single FMA
        float cb = fmaf(mval, cnew - cold, cold);
        c[sidx]     = cb;
        hf_w[sidx]  = hb;
        hbf_w[sidx] = (__bf16)hb;
        out[((size_t)brow * T_ + t) * (2 * H_) + dir * H_ + j] = hb;
    }
}

__global__ void finalize_kernel(const float* __restrict__ hf_fin,
                                const float* __restrict__ c,
                                float* __restrict__ ht_out,
                                float* __restrict__ ct_out) {
    int idx = blockIdx.x * blockDim.x + threadIdx.x;   // 2*B*H
    if (idx >= 2 * B_ * H_) return;
    int j   = idx & 511;
    int b   = (idx >> 9) & 63;
    int dir = idx >> 15;
    size_t sidx = ((size_t)dir * B_ + b) * H_ + j;
    ht_out[b * (2 * H_) + dir * H_ + j] = hf_fin[sidx];
    ct_out[b * (2 * H_) + dir * H_ + j] = c[sidx];
}

// ---------------- host ----------------

extern "C" void kernel_launch(void* const* d_in, const int* in_sizes, int n_in,
                              void* d_out, int out_size, void* d_ws, size_t ws_size,
                              hipStream_t stream) {
    const int*   inputs  = (const int*)d_in[0];
    const int*   lengths = (const int*)d_in[1];
    const float* emb     = (const float*)d_in[2];
    const float* Wih_f   = (const float*)d_in[3];
    const float* Whh_f   = (const float*)d_in[4];
    const float* bih_f   = (const float*)d_in[5];
    const float* bhh_f   = (const float*)d_in[6];
    const float* Wih_b   = (const float*)d_in[7];
    const float* Whh_b   = (const float*)d_in[8];
    const float* bih_b   = (const float*)d_in[9];
    const float* bhh_b   = (const float*)d_in[10];

    char* wsp = (char*)d_ws;
    auto alloc = [&](size_t bytes) -> void* {
        void* p = (void*)wsp;
        wsp += (bytes + 255) & ~(size_t)255;
        return p;
    };
    __bf16* X      = (__bf16*)alloc((size_t)T_ * B_ * E_ * sizeof(__bf16));
    float*  maskT  = (float*) alloc((size_t)T_ * B_ * sizeof(float));
    __bf16* Wc     = (__bf16*)alloc((size_t)2 * G4_ * KTOT * sizeof(__bf16));
    float*  biasc  = (float*) alloc((size_t)2 * G4_ * sizeof(float));
    __bf16* hbf    = (__bf16*)alloc((size_t)2 * 2 * B_ * H_ * sizeof(__bf16)); // [buf][dir][B][H]
    float*  hf     = (float*) alloc((size_t)2 * 2 * B_ * H_ * sizeof(float));  // [buf][dir][B][H]
    float*  cst    = (float*) alloc((size_t)2 * B_ * H_ * sizeof(float));      // [dir][B][H]

    float* out      = (float*)d_out;                           // (B,T,2H)
    float* ht_out   = out + (size_t)B_ * T_ * 2 * H_;
    float* ct_out   = ht_out + (size_t)B_ * 2 * H_;
    float* mask_out = ct_out + (size_t)B_ * 2 * H_;

    const int TPB = 256;
    // prep
    embed_kernel<<<(T_ * B_ * (E_ / 8)) / TPB, TPB, 0, stream>>>(inputs, emb, X);
    mask_kernel<<<(B_ * T_) / TPB, TPB, 0, stream>>>(lengths, maskT, mask_out);
    wcomb_kernel<<<(int)(((size_t)2 * G4_ * KTOT) / TPB), TPB, 0, stream>>>(
        Wih_f, Whh_f, Wih_b, Whh_b, Wc);
    bias_kernel<<<(2 * G4_ + TPB - 1) / TPB, TPB, 0, stream>>>(bih_f, bhh_f, bih_b, bhh_b, biasc);
    // zero initial states (both h buffers, both dirs, and c)
    zero_bf16<<<(2 * 2 * B_ * H_) / TPB, TPB, 0, stream>>>(hbf, 2 * 2 * B_ * H_);
    zero_f32 <<<(2 * 2 * B_ * H_) / TPB, TPB, 0, stream>>>(hf, 2 * 2 * B_ * H_);
    zero_f32 <<<(2 * B_ * H_) / TPB, TPB, 0, stream>>>(cst, 2 * B_ * H_);

    const size_t SB = (size_t)2 * B_ * H_;   // per-buffer state element count
    int buf = 0;
    for (int s = 0; s < T_; ++s) {
        lstm_step_kernel<<<dim3(16, 2), TPB, 0, stream>>>(
            s, X, maskT, Wc, biasc,
            hbf + (size_t)buf * SB, hbf + (size_t)(1 - buf) * SB,
            hf  + (size_t)buf * SB, hf  + (size_t)(1 - buf) * SB,
            cst, out);
        buf ^= 1;   // 'buf' now indexes the freshly written state
    }
    finalize_kernel<<<(2 * B_ * H_) / TPB, TPB, 0, stream>>>(
        hf + (size_t)buf * SB, cst, ht_out, ct_out);
}
